// Final_EEG_VQ_VAE_84748294685322
// MI455X (gfx1250) — matrix operand order repaired and checked by reading, not compiled, not run on hardware
//
#include <hip/hip_runtime.h>
#include <hip/hip_bf16.h>
#include <math.h>

// ---------------------------------------------------------------------------
// VQ-VAE encoder forward for MI455X (gfx1250, wave32, WMMA bf16 + TDM).
//   k_prep      : bf16 images: W1^T, W2^T, W3^T, codebook (for TDM + b128 LDS)
//   k_init      : zero BN stat accumulators
//   k_gemm1     : h1 = x@W1+b1 (WMMA bf16) -> h1 bf16; col sum/sumsq atomics
//   k_bnfinal   : BN stats -> per-column scale/shift
//   k_cnorm     : ||codebook_k||^2
//   k_gemm2     : h2 = relu( bn_relu(h1) @ W2 + b2 )  (TDM-staged W2 halves)
//   k_gemm3_vq  : z = tanh(h2@W3+b3); argmin_k ||e||^2 - 2 z.e ; idx + z_q
// Weights live in LDS transposed [N][Kld] with Kld = K+8 so B-fragments are
// two ds_load_b128 per lane and lane->bank stride is 4 (mod 64).
// ---------------------------------------------------------------------------

typedef __bf16 bf16;
typedef __attribute__((ext_vector_type(16))) __bf16 v16bf;
typedef __attribute__((ext_vector_type(8)))  __bf16 v8bf;
typedef __attribute__((ext_vector_type(8)))  float  v8f;
typedef __attribute__((ext_vector_type(4)))  unsigned int v4u;
typedef __attribute__((ext_vector_type(8)))  int v8i;
typedef __attribute__((ext_vector_type(4)))  int v4i;

#define B_N   65536
#define D_IN  128
#define H_N   256
#define D_E   64
#define K_N   4096
#define BN_EPS 1e-5f

__device__ __forceinline__ v8f zero8() {
  v8f z;
#pragma unroll
  for (int i = 0; i < 8; ++i) z[i] = 0.f;
  return z;
}

__device__ __forceinline__ v8f wmma_bf16(v16bf a, v16bf b, v8f c) {
  return __builtin_amdgcn_wmma_f32_16x16x32_bf16(false, a, false, b, (short)0, c,
                                                 false, false);
}

__device__ __forceinline__ v16bf combine16(v8bf lo, v8bf hi) {
  return __builtin_shufflevector(lo, hi, 0, 1, 2, 3, 4, 5, 6, 7, 8, 9, 10, 11,
                                 12, 13, 14, 15);
}

// ---------------------------------------------------------------------------
// Tensor Data Mover: 2D tiled load Global->LDS with optional LDS row padding.
// D# packing per CDNA5 ISA (group0 + group1, groups 2/3=0).
// 6-arg builtin form (clang-23 / therock-10.0 headers).
__device__ __forceinline__ void tdm_load_2d(unsigned lds_off, const void* src,
                                            unsigned rowlen, unsigned nrows,
                                            unsigned stride, unsigned dsize_code,
                                            unsigned pad_en, unsigned pad_iv,
                                            unsigned pad_amt) {
  unsigned long long ga = (unsigned long long)(size_t)src;
  v4u g0;
  g0[0] = 1u;                                   // count=1 (valid descriptor)
  g0[1] = lds_off;                              // lds_addr (bytes)
  g0[2] = (unsigned)ga;                         // global_addr[31:0]
  g0[3] = (unsigned)((ga >> 32) & 0x1FFFFFFu) | 0x80000000u;  // addr hi | type=2
  v8i g1;
  g1[0] = (int)((dsize_code << 16) | (pad_en << 20) | (pad_iv << 22) |
                (pad_amt << 25));               // mask=0, flags
  g1[1] = (int)(rowlen << 16);                  // tensor_dim0[15:0] @ 63:48
  g1[2] = (int)((rowlen >> 16) | (nrows << 16));   // dim0 hi | dim1 lo
  g1[3] = (int)((nrows >> 16) | (rowlen << 16));   // dim1 hi | tile_dim0
  g1[4] = (int)(nrows & 0xFFFFu);               // tile_dim1 (tile_dim2=0)
  g1[5] = (int)stride;                          // tensor_dim0_stride[31:0]
  g1[6] = 0;                                    // stride hi | dim1_stride lo
  g1[7] = 0;
  v4i z4 = {0, 0, 0, 0};
  v8i z8 = {0, 0, 0, 0, 0, 0, 0, 0};
  __builtin_amdgcn_tensor_load_to_lds(g0, g1, z4, z4, z8, 0);
}

__device__ __forceinline__ unsigned lds_offset_of(const void* p) {
  return (unsigned)(size_t)p;   // generic LDS pointer: low 32 bits = LDS offset
}

// ---------------------------------------------------------------------------
// A fragment (16x32 bf16) from row-major fp32; kb = k0 + (lane>>4)*8.
__device__ __forceinline__ v16bf a_frag_f32(const float* __restrict__ src, int ld,
                                            int m, int kb) {
  const float4* p = reinterpret_cast<const float4*>(src + (size_t)m * ld + kb);
  float4 q0 = p[0], q1 = p[1], q2 = p[4], q3 = p[5];
  v16bf a;
  a[0]=(bf16)q0.x;  a[1]=(bf16)q0.y;  a[2]=(bf16)q0.z;  a[3]=(bf16)q0.w;
  a[4]=(bf16)q1.x;  a[5]=(bf16)q1.y;  a[6]=(bf16)q1.z;  a[7]=(bf16)q1.w;
  a[8]=(bf16)q2.x;  a[9]=(bf16)q2.y;  a[10]=(bf16)q2.z; a[11]=(bf16)q2.w;
  a[12]=(bf16)q3.x; a[13]=(bf16)q3.y; a[14]=(bf16)q3.z; a[15]=(bf16)q3.w;
  return a;
}

// A fragment from row-major bf16 (two b128 loads).
__device__ __forceinline__ v16bf a_frag_bf16(const bf16* __restrict__ src, int ld,
                                             int m, int kb) {
  const bf16* p = src + (size_t)m * ld + kb;
  v8bf lo = *reinterpret_cast<const v8bf*>(p);
  v8bf hi = *reinterpret_cast<const v8bf*>(p + 16);
  return combine16(lo, hi);
}

// A fragment from h1 bf16 with fused BatchNorm + ReLU (fp32 math).
__device__ __forceinline__ v16bf a_frag_bn(const bf16* __restrict__ src, int m,
                                           int kb, const float* sScale,
                                           const float* sShift) {
  const bf16* p = src + (size_t)m * H_N + kb;
  v8bf lo = *reinterpret_cast<const v8bf*>(p);
  v8bf hi = *reinterpret_cast<const v8bf*>(p + 16);
  v16bf a;
#pragma unroll
  for (int i = 0; i < 8; ++i) {
    int ka = kb + i, kc = kb + 16 + i;
    a[i]     = (bf16)fmaxf((float)lo[i] * sScale[ka] + sShift[ka], 0.f);
    a[i + 8] = (bf16)fmaxf((float)hi[i] * sScale[kc] + sShift[kc], 0.f);
  }
  return a;
}

// B fragment (32x16) from transposed LDS tile [N][kld] -> 2x ds_load_b128.
__device__ __forceinline__ v16bf b_frag_ldsT(const bf16* w, int kld, int k0,
                                             int n0, int lane) {
  const bf16* p = w + (n0 + (lane & 15)) * kld + k0 + ((lane >> 4) << 3);
  v8bf lo = *reinterpret_cast<const v8bf*>(p);
  v8bf hi = *reinterpret_cast<const v8bf*>(p + 16);
  return combine16(lo, hi);
}

// B fragment from bf16 codebook: B[k][n] = cb[n0+n][k] (per-lane contiguous).
__device__ __forceinline__ v16bf b_frag_cb(const bf16* __restrict__ cb, int n0,
                                           int ks, int lane) {
  const bf16* p =
      cb + (size_t)(n0 + (lane & 15)) * D_E + ks * 32 + ((lane >> 4) << 3);
  v8bf lo = *reinterpret_cast<const v8bf*>(p);
  v8bf hi = *reinterpret_cast<const v8bf*>(p + 16);
  return combine16(lo, hi);
}

// ---------------------------------------------------------------------------
__global__ void k_prep(const float* __restrict__ W1, const float* __restrict__ W2,
                       const float* __restrict__ W3, const float* __restrict__ cb,
                       bf16* __restrict__ w1t, bf16* __restrict__ w2t,
                       bf16* __restrict__ w3t, bf16* __restrict__ cbb) {
  int t = blockIdx.x * 256 + threadIdx.x;            // grid 1024 -> t < 262144
  cbb[t] = (bf16)cb[t];
  if (t < 65536) { int n = t >> 8, k = t & 255; w2t[t] = (bf16)W2[k * 256 + n]; }
  if (t < 32768) { int n = t >> 7, k = t & 127; w1t[t] = (bf16)W1[k * 256 + n]; }
  if (t < 16384) { int n = t >> 8, k = t & 255; w3t[t] = (bf16)W3[k * 64 + n]; }
}

// ---------------------------------------------------------------------------
__global__ void k_init(float* gSum, float* gSq) {
  int t = threadIdx.x;
  gSum[t] = 0.f;
  gSq[t]  = 0.f;
}

// ---------------------------------------------------------------------------
#define KLD1 (D_IN + 8)    // 136
__global__ __launch_bounds__(256) void k_gemm1(const float* __restrict__ x,
                                               const bf16* __restrict__ w1t,
                                               const float* __restrict__ b1,
                                               bf16* __restrict__ h1,
                                               float* __restrict__ gSum,
                                               float* __restrict__ gSq) {
  __shared__ __align__(16) bf16 sW1[H_N * KLD1];     // 256 x 136 bf16 = 68 KB
  __shared__ float sSum[H_N];
  __shared__ float sSq[H_N];
  int tid = threadIdx.x;
  int lane = tid & 31, wave = tid >> 5;
  sSum[tid] = 0.f;
  sSq[tid]  = 0.f;
  if (wave == 0) {
    // W1^T [256 rows][128 elems], pad each 64-dword row by 4 dwords -> Kld 136.
    tdm_load_2d(lds_offset_of(sW1), w1t, D_IN, H_N, D_IN, /*dsize 2B*/ 1,
                /*pad*/ 1, /*iv 64dw*/ 5, /*amt 4dw*/ 3);
    __builtin_amdgcn_s_wait_tensorcnt(0);
  }
  __syncthreads();

  int hi = lane >> 4, l4 = lane & 15;
  int m0 = blockIdx.x * 128 + wave * 16;
  int m  = m0 + l4;

  v16bf af[4];
#pragma unroll
  for (int ks = 0; ks < 4; ++ks)
    af[ks] = a_frag_f32(x, D_IN, m, ks * 32 + (hi << 3));

#pragma unroll
  for (int ct = 0; ct < 16; ++ct) {
    int n0 = ct * 16;
    v8f acc = zero8();
#pragma unroll
    for (int ks = 0; ks < 4; ++ks)
      acc = wmma_bf16(af[ks], b_frag_ldsT(sW1, KLD1, ks * 32, n0, lane), acc);
    int   n    = n0 + l4;
    float bias = b1[n];
    float s = 0.f, q = 0.f;
#pragma unroll
    for (int v = 0; v < 8; ++v) {
      float h = acc[v] + bias;
      s += h; q += h * h;
      int row = m0 + v + 8 * hi;
      h1[(size_t)row * H_N + n] = (bf16)h;
    }
    atomicAdd(&sSum[n], s);
    atomicAdd(&sSq[n], q);
  }
  __syncthreads();
  atomicAdd(&gSum[tid], sSum[tid]);
  atomicAdd(&gSq[tid],  sSq[tid]);
}

// ---------------------------------------------------------------------------
__global__ void k_bnfinal(const float* gSum, const float* gSq,
                          const float* gamma, const float* beta,
                          float* scale, float* shift) {
  int t = threadIdx.x;
  float mu  = gSum[t] * (1.f / (float)B_N);
  float var = gSq[t] * (1.f / (float)B_N) - mu * mu;   // biased variance
  float sc  = gamma[t] * rsqrtf(var + BN_EPS);
  scale[t] = sc;
  shift[t] = beta[t] - mu * sc;
}

// ---------------------------------------------------------------------------
__global__ void k_cnorm(const float* __restrict__ cb, float* __restrict__ cnorm) {
  int c = blockIdx.x * 256 + threadIdx.x;
  const float4* p = reinterpret_cast<const float4*>(cb + (size_t)c * D_E);
  float s = 0.f;
#pragma unroll
  for (int i = 0; i < 16; ++i) {
    float4 q = p[i];
    s += q.x * q.x + q.y * q.y + q.z * q.z + q.w * q.w;
  }
  cnorm[c] = s;
}

// ---------------------------------------------------------------------------
#define KLD2 (128 + 8)     // 136 per k-phase
__global__ __launch_bounds__(256) void k_gemm2(const bf16* __restrict__ h1,
                                               const bf16* __restrict__ w2t,
                                               const float* __restrict__ b2,
                                               const float* __restrict__ scale,
                                               const float* __restrict__ shift,
                                               bf16* __restrict__ h2) {
  __shared__ __align__(16) bf16 sW2[H_N * KLD2];     // 68 KB (half of W2)
  __shared__ float sScale[H_N], sShift[H_N], sB2[H_N];
  int tid = threadIdx.x;
  sScale[tid] = scale[tid];
  sShift[tid] = shift[tid];
  sB2[tid]    = b2[tid];

  int lane = tid & 31, wave = tid >> 5;
  int hi = lane >> 4, l4 = lane & 15;
  int m0 = blockIdx.x * 128 + wave * 16;
  int m  = m0 + l4;

  v8f acc[16];
#pragma unroll
  for (int ct = 0; ct < 16; ++ct) acc[ct] = zero8();

  for (int p = 0; p < 2; ++p) {          // k-phases: K = [0,128), [128,256)
    __syncthreads();                     // prior phase reads complete
    if (wave == 0) {
      // W2^T rows n=0..255, k-slice [p*128, p*128+128), source row stride 256.
      tdm_load_2d(lds_offset_of(sW2), w2t + p * 128, 128, H_N, H_N, 1,
                  1, 5, 3);
      __builtin_amdgcn_s_wait_tensorcnt(0);
    }
    __syncthreads();

    v16bf af[4];
#pragma unroll
    for (int ks = 0; ks < 4; ++ks)
      af[ks] = a_frag_bn(h1, m, p * 128 + ks * 32 + (hi << 3), sScale, sShift);

#pragma unroll
    for (int ct = 0; ct < 16; ++ct) {
#pragma unroll
      for (int ks = 0; ks < 4; ++ks)
        acc[ct] = wmma_bf16(af[ks],
                            b_frag_ldsT(sW2, KLD2, ks * 32, ct * 16, lane),
                            acc[ct]);
    }
  }

#pragma unroll
  for (int ct = 0; ct < 16; ++ct) {
    int   n    = ct * 16 + l4;
    float bias = sB2[n];
#pragma unroll
    for (int v = 0; v < 8; ++v) {
      float h = fmaxf(acc[ct][v] + bias, 0.f);
      int row = m0 + v + 8 * hi;
      h2[(size_t)row * H_N + n] = (bf16)h;
    }
  }
}

// ---------------------------------------------------------------------------
#define KLD3 (H_N + 8)     // 264
#define ZLD  (D_E + 8)     // 72
__global__ __launch_bounds__(256) void k_gemm3_vq(const bf16* __restrict__ h2,
                                                  const bf16* __restrict__ w3t,
                                                  const float* __restrict__ b3,
                                                  const bf16* __restrict__ cbb,
                                                  const float* __restrict__ codebook,
                                                  const float* __restrict__ cnorm,
                                                  float* __restrict__ out_idx,
                                                  float* __restrict__ out_zq) {
  __shared__ __align__(16) bf16  sW3[D_E * KLD3];    // 64 x 264 bf16 = 33 KB
  __shared__ __align__(16) bf16  sZ[8 * 16 * ZLD];   // 18 KB per-wave staging
  __shared__ __align__(16) float sCn[K_N];           // 16 KB
  __shared__ float sB3[D_E];
  int tid = threadIdx.x;
  int lane = tid & 31, wave = tid >> 5;
  if (tid < D_E) sB3[tid] = b3[tid];
  if (wave == 0) {
    // W3^T [64 rows][256 elems]: 128-dword rows padded by 4 dwords -> Kld 264.
    tdm_load_2d(lds_offset_of(sW3), w3t, H_N, D_E, H_N, 1, 1, /*iv 128dw*/ 6, 3);
    // ||e||^2 table: flat 4096 fp32, no padding.
    tdm_load_2d(lds_offset_of(sCn), cnorm, K_N, 1, K_N, /*dsize 4B*/ 2, 0, 0, 0);
    __builtin_amdgcn_s_wait_tensorcnt(0);
  }
  __syncthreads();

  int hi = lane >> 4, l4 = lane & 15;
  int m0 = blockIdx.x * 128 + wave * 16;
  int m  = m0 + l4;

  // ---- GEMM3: z = tanh(h2 @ W3 + b3) --------------------------------------
  v16bf af[8];
#pragma unroll
  for (int ks = 0; ks < 8; ++ks)
    af[ks] = a_frag_bf16(h2, H_N, m, ks * 32 + (hi << 3));

  bf16* zbuf = sZ + wave * 16 * ZLD;     // wave-private, no barrier needed
#pragma unroll
  for (int ct = 0; ct < 4; ++ct) {
    int n0 = ct * 16;
    v8f acc = zero8();
#pragma unroll
    for (int ks = 0; ks < 8; ++ks)
      acc = wmma_bf16(af[ks], b_frag_ldsT(sW3, KLD3, ks * 32, n0, lane), acc);
    int   n    = n0 + l4;
    float bias = sB3[n];
#pragma unroll
    for (int v = 0; v < 8; ++v) {
      float z  = tanhf(acc[v] + bias);
      int   ml = v + 8 * hi;
      zbuf[ml * ZLD + n] = (bf16)z;
    }
  }

  // ---- VQ: argmin_k ||e_k||^2 - 2 z.e_k  (||z||^2 argmin-invariant) -------
  v16bf za[2];
#pragma unroll
  for (int ks = 0; ks < 2; ++ks) {
    int k0 = ks * 32 + (hi << 3);
    const bf16* p = zbuf + l4 * ZLD + k0;
    v8bf lo = *reinterpret_cast<const v8bf*>(p);
    v8bf hh = *reinterpret_cast<const v8bf*>(p + 16);
    za[ks] = combine16(lo, hh);
  }

  float best[8];
  int   bidx[8];
#pragma unroll
  for (int v = 0; v < 8; ++v) { best[v] = 3.0e38f; bidx[v] = 0; }

  for (int n0 = 0; n0 < K_N; n0 += 16) {
    __builtin_prefetch(cbb + (size_t)(n0 + 128) * D_E, 0, 0);
    v8f acc = zero8();
#pragma unroll
    for (int ks = 0; ks < 2; ++ks)
      acc = wmma_bf16(za[ks], b_frag_cb(cbb, n0, ks, lane), acc);
    float cn   = sCn[n0 + l4];
    int   code = n0 + l4;
#pragma unroll
    for (int v = 0; v < 8; ++v) {
      float score = cn - 2.f * acc[v];
      if (score < best[v]) { best[v] = score; bidx[v] = code; }
    }
  }

  // Cross-lane argmin within each 16-lane group (same row set), wave32-safe.
#pragma unroll
  for (int v = 0; v < 8; ++v) {
    float bs = best[v];
    int   bi = bidx[v];
#pragma unroll
    for (int off = 8; off > 0; off >>= 1) {
      float os = __shfl_xor(bs, off);
      int   oi = __shfl_xor(bi, off);
      if (os < bs || (os == bs && oi < bi)) { bs = os; bi = oi; }
    }
    int row = m0 + v + 8 * hi;
    if (l4 == 0) out_idx[row] = (float)bi;
    const float4* src =
        reinterpret_cast<const float4*>(codebook + (size_t)bi * D_E);
    reinterpret_cast<float4*>(out_zq + (size_t)row * D_E)[l4] = src[l4];
  }
}

// ---------------------------------------------------------------------------
extern "C" void kernel_launch(void* const* d_in, const int* in_sizes, int n_in,
                              void* d_out, int out_size, void* d_ws, size_t ws_size,
                              hipStream_t stream) {
  const float* x        = (const float*)d_in[0];
  const float* W1       = (const float*)d_in[1];
  const float* b1       = (const float*)d_in[2];
  const float* gamma    = (const float*)d_in[3];
  const float* beta     = (const float*)d_in[4];
  const float* W2       = (const float*)d_in[5];
  const float* b2       = (const float*)d_in[6];
  const float* W3       = (const float*)d_in[7];
  const float* b3       = (const float*)d_in[8];
  const float* codebook = (const float*)d_in[9];

  char*  ws    = (char*)d_ws;
  float* gSum  = (float*)(ws + 0);
  float* gSq   = (float*)(ws + 1024);
  float* scale = (float*)(ws + 2048);
  float* shift = (float*)(ws + 3072);
  float* cnorm = (float*)(ws + 4096);             // 16 KB
  bf16*  w1t   = (bf16*)(ws + 20480);             // 64 KB  (W1^T bf16)
  bf16*  w2t   = (bf16*)(ws + 86016);             // 128 KB (W2^T bf16)
  bf16*  w3t   = (bf16*)(ws + 217088);            // 32 KB  (W3^T bf16)
  bf16*  cbb   = (bf16*)(ws + 249856);            // 512 KB (codebook bf16)
  bf16*  h1    = (bf16*)(ws + 774144);            // 32 MB
  bf16*  h2    = (bf16*)(ws + 774144 + (size_t)B_N * H_N * 2);  // 32 MB

  float* out_idx = (float*)d_out;                 // [B] indices (as float)
  float* out_zq  = out_idx + B_N;                 // [B, D_e]

  k_prep<<<(K_N * D_E) / 256, 256, 0, stream>>>(W1, W2, W3, codebook,
                                                w1t, w2t, w3t, cbb);
  k_init<<<1, 256, 0, stream>>>(gSum, gSq);
  k_gemm1<<<B_N / 128, 256, 0, stream>>>(x, w1t, b1, h1, gSum, gSq);
  k_bnfinal<<<1, 256, 0, stream>>>(gSum, gSq, gamma, beta, scale, shift);
  k_cnorm<<<K_N / 256, 256, 0, stream>>>(codebook, cnorm);
  k_gemm2<<<B_N / 128, 256, 0, stream>>>(h1, w2t, b2, scale, shift, h2);
  k_gemm3_vq<<<B_N / 128, 256, 0, stream>>>(h2, w3t, b3, cbb, codebook, cnorm,
                                            out_idx, out_zq);
}
